// SeNaTraStage_37220186587448
// MI455X (gfx1250) — compile-verified, matrix operand order checked by reading, not run on hardware
//
#include <hip/hip_runtime.h>
#include <math.h>

typedef __attribute__((ext_vector_type(16))) _Float16 v16h;
typedef __attribute__((ext_vector_type(8)))  _Float16 v8h;
typedef __attribute__((ext_vector_type(2)))  _Float16 h2;
typedef __attribute__((ext_vector_type(8)))  float    v8f;
typedef _Float16 half_t;
typedef unsigned int u32;

#define B_      16
#define H_      56
#define HO_     28
#define NIN_    3136      // 56*56
#define NOUT_   784       // 28*28
#define D_      96
#define DO_     192
#define HID_    288
#define BHID_   576
#define MROWS_  12544     // B*NOUT
#define MIN_    50176     // B*NIN

__device__ __forceinline__ int iabs_(int a) { return a < 0 ? -a : a; }

// ============================================================================
// WMMA GEMM: out = act(A[M,K](f16) @ W[K,N](packed f16 pairs) + bias[N]) (+res)
// Wp layout: dword at [(k/2)*N + n] holds halves (k, k+1) of column n.
// block = 256 (8 waves); wave computes 16 rows x 48 cols (3 wmma tiles).
// grid = (M/128, N/48). Requires M%128==0, N%48==0, K%32==0.
//
// B tile (32k x 48n, identical for all 8 waves) is staged in LDS in fragment
// order [tile][lane][vgpr] with double buffering: cooperative coalesced fill,
// then each wave reads its v16h B fragment as one aligned 32B LDS load.
// A fragment: two aligned v8h (16B) global loads. No cvt anywhere in the loop.
// ============================================================================
__device__ __forceinline__ void fillB(u32* __restrict__ dst, const u32* __restrict__ Wp,
                                      int kk0, int N, int nbase, int tid)
{
#pragma unroll
    for (int i = 0; i < 3; ++i) {
        int e = tid + 256 * i;        // 0..767 : 16 kk-rows x 48 cols
        int kkrow = e / 48;
        int ncol  = e % 48;
        u32 d = Wp[(size_t)(kk0 + kkrow) * N + nbase + ncol];
        // fragment position: kk = (v&3) + h*4 + (v>>2)*8
        int va = kkrow & 3, hh = (kkrow >> 2) & 1, vb = kkrow >> 3;
        int v  = va | (vb << 2);
        int l  = hh * 16 + (ncol & 15);
        int t  = ncol >> 4;
        dst[t * 256 + l * 8 + v] = d;
    }
}

__global__ __launch_bounds__(256)
void wmma_gemm(const half_t* __restrict__ A, const u32* __restrict__ Wp,
               const float* __restrict__ bias, const float* __restrict__ res,
               float* __restrict__ C, half_t* __restrict__ C16,
               int M, int N, int K, int gelu)
{
    __shared__ u32 ldsB[2][768];      // 6 KB double buffer

    const int tid  = threadIdx.x;
    const int lane = tid & 31;
    const int wave = tid >> 5;
    const int h    = lane >> 4;       // lane half
    const int l16  = lane & 15;
    const int mbase = (blockIdx.x * 8 + wave) * 16;
    const int nbase = blockIdx.y * 48;

    v8f acc[3] = {};
    const half_t* Arow = A + (size_t)(mbase + l16) * K;
    const int nsteps = K >> 5;

    fillB(ldsB[0], Wp, 0, N, nbase, tid);

    for (int s = 0; s < nsteps; ++s) {
        __syncthreads();              // buf[s&1] ready; prior reads of it done
        const int kb = s << 5;

        // A fragment 16x32 f16 (ISA 7.12.2): halves [h*8..h*8+7], [16+h*8..]
        const v8h* Ap = (const v8h*)(Arow + kb);
        v8h alo = Ap[h];
        v8h ahi = Ap[2 + h];
        v16h af;
#pragma unroll
        for (int i = 0; i < 8; ++i) { af[i] = alo[i]; af[8 + i] = ahi[i]; }
        if (s + 1 < nsteps) __builtin_prefetch(Arow + kb + 32, 0, 0);

        const u32* buf = ldsB[s & 1];
#pragma unroll
        for (int t = 0; t < 3; ++t) {
            v16h bf = *(const v16h*)(buf + t * 256 + lane * 8);  // 32B aligned
            acc[t] = __builtin_amdgcn_wmma_f32_16x16x32_f16(
                false, af, false, bf, (short)0, acc[t], false, false);
        }
        if (s + 1 < nsteps)
            fillB(ldsB[(s + 1) & 1], Wp, (s + 1) << 4, N, nbase, tid);
    }

    // C layout: VGPR r -> row r + 8*h ; col = l16
#pragma unroll
    for (int t = 0; t < 3; ++t) {
        int ncol = nbase + t * 16 + l16;
        float bs = bias[ncol];
#pragma unroll
        for (int r = 0; r < 8; ++r) {
            int mrow = mbase + r + 8 * h;
            float val = acc[t][r] + bs;
            if (gelu) val = 0.5f * val * (1.0f + erff(val * 0.70710678f));
            size_t idx = (size_t)mrow * N + ncol;
            if (res) val += res[idx];
            if (C)   C[idx] = val;
            if (C16) C16[idx] = (half_t)val;
        }
    }
}

// ============================================================================
// LayerNorm: y = ln(X[row])*w + b (+ addto[row]); writes f32 out and/or f16.
// Wave per row (wave32), butterfly shuffle reductions.
// ============================================================================
template <int DD>
__global__ __launch_bounds__(256)
void ln_kernel(const float* __restrict__ X, const float* __restrict__ w,
               const float* __restrict__ b, const float* __restrict__ addto,
               float* __restrict__ out, half_t* __restrict__ out16, int rows)
{
    constexpr int NPER = DD / 32;
    const int lane = threadIdx.x & 31;
    const int row  = blockIdx.x * 8 + (threadIdx.x >> 5);
    if (row >= rows) return;
    const float* x = X + (size_t)row * DD;
    float v[NPER], s = 0.f, s2 = 0.f;
#pragma unroll
    for (int t = 0; t < NPER; ++t) {
        float u = x[lane + 32 * t];
        v[t] = u; s += u; s2 += u * u;
    }
#pragma unroll
    for (int m = 16; m; m >>= 1) { s += __shfl_xor(s, m, 32); s2 += __shfl_xor(s2, m, 32); }
    float mean = s / DD;
    float rstd = rsqrtf(s2 / DD - mean * mean + 1e-5f);
    const float* a = addto ? addto + (size_t)row * DD : nullptr;
#pragma unroll
    for (int t = 0; t < NPER; ++t) {
        int d = lane + 32 * t;
        float y = (v[t] - mean) * rstd * w[d] + b[d];
        if (a) y += a[d];
        if (out)   out[(size_t)row * DD + d] = y;
        if (out16) out16[(size_t)row * DD + d] = (half_t)y;
    }
}

// ============================================================================
// Grouping attention: wave per output token; 9-neighbor window on 56x56 grid
// (locality-mask sparsity). Writes nc; last iter scatters A_ups into d_out.
// ============================================================================
__global__ __launch_bounds__(256)
void group_attn(const float* __restrict__ Qo, const float* __restrict__ Kin,
                const float* __restrict__ Vin, const float* __restrict__ rel_bias,
                float* __restrict__ nc, float* __restrict__ aups, int writeA)
{
    const int lane = threadIdx.x & 31;
    const int o    = blockIdx.x * 8 + (threadIdx.x >> 5);   // 0..MROWS-1
    const int b = o / NOUT_, ol = o % NOUT_;
    const int io = ol / HO_, jo = ol % HO_;

    // bias = mean(rel_bias[25])
    float bsum = (lane < 25) ? rel_bias[lane] : 0.f;
#pragma unroll
    for (int m = 16; m; m >>= 1) bsum += __shfl_xor(bsum, m, 32);
    const float bias = bsum * (1.0f / 25.0f);

    const float* q = Qo + (size_t)o * D_;
    float q0 = q[lane], q1 = q[lane + 32], q2 = q[lane + 64];

    float p[9]; int nn[9]; float mx = -1e30f;
#pragma unroll
    for (int j = 0; j < 9; ++j) {
        int yi = 2 * io + (j / 3) - 1;
        int xi = 2 * jo + (j % 3) - 1;
        bool ok = (yi >= 0) && (yi < H_) && (xi >= 0) && (xi < H_);  // wave-uniform
        nn[j] = ok ? (b * NIN_ + yi * H_ + xi) : -1;
        float part = 0.f;
        if (ok) {
            const float* kr = Kin + (size_t)nn[j] * D_;
            part = q0 * kr[lane] + q1 * kr[lane + 32] + q2 * kr[lane + 64];
        }
#pragma unroll
        for (int m = 16; m; m >>= 1) part += __shfl_xor(part, m, 32);
        p[j] = ok ? (part + bias) : -1e30f;
        mx = fmaxf(mx, p[j]);
    }
    float den = 0.f;
#pragma unroll
    for (int j = 0; j < 9; ++j) { p[j] = (nn[j] >= 0) ? expf(p[j] - mx) : 0.f; den += p[j]; }
    float inv = 1.0f / den;
    float a0 = 0.f, a1 = 0.f, a2 = 0.f;
#pragma unroll
    for (int j = 0; j < 9; ++j) {
        p[j] *= inv;
        if (nn[j] >= 0) {
            const float* vr = Vin + (size_t)nn[j] * D_;
            a0 += p[j] * vr[lane]; a1 += p[j] * vr[lane + 32]; a2 += p[j] * vr[lane + 64];
        }
    }
    float* no = nc + (size_t)o * D_;
    no[lane] = a0; no[lane + 32] = a1; no[lane + 64] = a2;
    if (writeA && lane == 0) {
#pragma unroll
        for (int j = 0; j < 9; ++j)
            if (nn[j] >= 0)
                aups[(size_t)o * NIN_ + (nn[j] - b * NIN_)] = p[j];
    }
}

// ============================================================================
// A_down = A_ups / (1e-10 + column sum). Thread per (b, n).
// ============================================================================
__global__ void adown_kernel(const float* __restrict__ aups, float* __restrict__ adown)
{
    int t = blockIdx.x * blockDim.x + threadIdx.x;
    if (t >= B_ * NIN_) return;
    int b = t / NIN_, nl = t % NIN_;
    int yi = nl / H_, xi = nl % H_;
    int io0 = yi / 2 - 1; if (io0 < 0) io0 = 0;
    int io1 = yi / 2 + 1; if (io1 > HO_ - 1) io1 = HO_ - 1;
    int jo0 = xi / 2 - 1; if (jo0 < 0) jo0 = 0;
    int jo1 = xi / 2 + 1; if (jo1 > HO_ - 1) jo1 = HO_ - 1;
    float s = 0.f;
    for (int io = io0; io <= io1; ++io) {
        if (iabs_(yi - 2 * io) >= 2) continue;
        for (int jo = jo0; jo <= jo1; ++jo) {
            if (iabs_(xi - 2 * jo) >= 2) continue;
            s += aups[(size_t)(b * NOUT_ + io * HO_ + jo) * NIN_ + nl];
        }
    }
    float inv = 1.0f / (1e-10f + s);
    for (int io = io0; io <= io1; ++io) {
        if (iabs_(yi - 2 * io) >= 2) continue;
        for (int jo = jo0; jo <= jo1; ++jo) {
            if (iabs_(xi - 2 * jo) >= 2) continue;
            size_t idx = (size_t)(b * NOUT_ + io * HO_ + jo) * NIN_ + nl;
            adown[idx] = aups[idx] * inv;
        }
    }
}

// ============================================================================
// Block 3x3 window attention (zero-padded k/v => score 0 / v 0, as reference).
// Wave per token; lane owns 6 of 192 dims. xout = xin + att (in-place safe).
// ============================================================================
__global__ __launch_bounds__(256)
void win_attn(const float* __restrict__ q, const float* __restrict__ k,
              const float* __restrict__ v, const float* __restrict__ xin,
              float* __restrict__ xout)
{
    const int lane = threadIdx.x & 31;
    const int o = blockIdx.x * 8 + (threadIdx.x >> 5);
    const int b = o / NOUT_, ol = o % NOUT_;
    const int ho = ol / HO_, wo = ol % HO_;
    const float scale = 0.07216878f;   // 192^-0.5
    const float* qr = q + (size_t)o * DO_;
    float qv[6];
#pragma unroll
    for (int t = 0; t < 6; ++t) qv[t] = qr[lane + 32 * t];
    float p[9]; int nn[9]; float mx = -1e30f;
#pragma unroll
    for (int j = 0; j < 9; ++j) {
        int y = ho + j / 3 - 1, x = wo + j % 3 - 1;
        bool ok = (y >= 0) && (y < HO_) && (x >= 0) && (x < HO_);
        nn[j] = ok ? (b * NOUT_ + y * HO_ + x) : -1;
        float part = 0.f;
        if (ok) {
            const float* kr = k + (size_t)nn[j] * DO_;
#pragma unroll
            for (int t = 0; t < 6; ++t) part += qv[t] * kr[lane + 32 * t];
        }
#pragma unroll
        for (int m = 16; m; m >>= 1) part += __shfl_xor(part, m, 32);
        p[j] = ok ? part * scale : 0.0f;
        mx = fmaxf(mx, p[j]);
    }
    float den = 0.f;
#pragma unroll
    for (int j = 0; j < 9; ++j) { p[j] = expf(p[j] - mx); den += p[j]; }
    float inv = 1.0f / den;
    float acc[6] = {};
#pragma unroll
    for (int j = 0; j < 9; ++j) {
        float pj = p[j] * inv;
        if (nn[j] >= 0) {
            const float* vr = v + (size_t)nn[j] * DO_;
#pragma unroll
            for (int t = 0; t < 6; ++t) acc[t] += pj * vr[lane + 32 * t];
        }
    }
    float* outr = xout + (size_t)o * DO_;
    const float* inr = xin + (size_t)o * DO_;
#pragma unroll
    for (int t = 0; t < 6; ++t) outr[lane + 32 * t] = inr[lane + 32 * t] + acc[t];
}

// ============================================================================
// Helpers
// ============================================================================
__global__ void pack_patches(const float* __restrict__ x, half_t* __restrict__ P)
{
    int idx = blockIdx.x * blockDim.x + threadIdx.x;
    if (idx >= MROWS_ * 384) return;
    int row = idx / 384, col = idx % 384;
    int b = row / NOUT_, ol = row % NOUT_;
    int io = ol / HO_, jo = ol % HO_;
    int i = col / 192, j = (col / 96) & 1, c = col % 96;
    P[idx] = (half_t)x[((size_t)b * NIN_ + (size_t)(2 * io + i) * H_ + (2 * jo + j)) * D_ + c];
}

__global__ void repack_conv(const float* __restrict__ cw, float* __restrict__ Wc)
{
    int idx = blockIdx.x * blockDim.x + threadIdx.x;
    if (idx >= 384 * 96) return;
    int col = idx / 96, o = idx % 96;
    int i = col / 192, j = (col / 96) & 1, c = col % 96;
    Wc[(size_t)col * 96 + o] = cw[((size_t)o * 96 + c) * 4 + i * 2 + j];
}

__global__ void pack_weight(const float* __restrict__ W, u32* __restrict__ Wp, int K, int N)
{
    int idx = blockIdx.x * blockDim.x + threadIdx.x;
    if (idx >= (K / 2) * N) return;
    int kk = idx / N, n = idx % N;
    h2 p;
    p[0] = (half_t)W[(size_t)(2 * kk) * N + n];
    p[1] = (half_t)W[(size_t)(2 * kk + 1) * N + n];
    Wp[idx] = __builtin_bit_cast(u32, p);
}

__global__ void cvt_half(const float* __restrict__ src, half_t* __restrict__ dst, size_t n)
{
    size_t i = (size_t)blockIdx.x * blockDim.x + threadIdx.x;
    size_t stride = (size_t)gridDim.x * blockDim.x;
    for (; i < n; i += stride) dst[i] = (half_t)src[i];
}

__global__ void zero_f(float* __restrict__ p, size_t n)
{
    size_t i = (size_t)blockIdx.x * blockDim.x + threadIdx.x;
    size_t stride = (size_t)gridDim.x * blockDim.x;
    for (; i < n; i += stride) p[i] = 0.0f;
}

// ============================================================================
extern "C" void kernel_launch(void* const* d_in, const int* in_sizes, int n_in,
                              void* d_out, int out_size, void* d_ws, size_t ws_size,
                              hipStream_t stream)
{
    const float* x        = (const float*)d_in[0];
    const float* conv_w   = (const float*)d_in[1];
    const float* conv_b   = (const float*)d_in[2];
    const float* ln_init_w= (const float*)d_in[3];
    const float* ln_init_b= (const float*)d_in[4];
    const float* q_w = (const float*)d_in[5];  const float* q_b = (const float*)d_in[6];
    const float* k_w = (const float*)d_in[7];  const float* k_b = (const float*)d_in[8];
    const float* v_w = (const float*)d_in[9];  const float* v_b = (const float*)d_in[10];
    const float* gm1_w = (const float*)d_in[11]; const float* gm1_b = (const float*)d_in[12];
    const float* gm2_w = (const float*)d_in[13]; const float* gm2_b = (const float*)d_in[14];
    const float* mlp_norm_w = (const float*)d_in[15]; const float* mlp_norm_b = (const float*)d_in[16];
    const float* ln_out_w = (const float*)d_in[17]; const float* ln_out_b = (const float*)d_in[18];
    const float* rel_bias = (const float*)d_in[19];
    const float* proj_w = (const float*)d_in[20]; const float* proj_b = (const float*)d_in[21];
    const float* norm_w = (const float*)d_in[22]; const float* norm_b = (const float*)d_in[23];
    const float* blk_ln1_w = (const float*)d_in[24]; const float* blk_ln1_b = (const float*)d_in[25];
    const float* blk_q_w = (const float*)d_in[26]; const float* blk_q_b = (const float*)d_in[27];
    const float* blk_k_w = (const float*)d_in[28]; const float* blk_k_b = (const float*)d_in[29];
    const float* blk_v_w = (const float*)d_in[30]; const float* blk_v_b = (const float*)d_in[31];
    const float* blk_ln2_w = (const float*)d_in[32]; const float* blk_ln2_b = (const float*)d_in[33];
    const float* blk_m1_w = (const float*)d_in[34]; const float* blk_m1_b = (const float*)d_in[35];
    const float* blk_m2_w = (const float*)d_in[36]; const float* blk_m2_b = (const float*)d_in[37];

    float* out_xb    = (float*)d_out;                         // (B, 784, 192)
    float* out_adown = out_xb + (size_t)MROWS_ * DO_;         // (B, 784, 3136)
    float* out_aups  = out_adown + (size_t)B_ * NOUT_ * NIN_; // (B, 784, 3136)

    // --- workspace layout (bump allocator, phase-based reuse) ---
    char* wsb = (char*)d_ws;
    size_t off = 0;
    auto alloc = [&](size_t bytes) { char* p = wsb + off; off += (bytes + 255) & ~(size_t)255; return (void*)p; };

    float*  WcT   = (float*) alloc(384 * 96 * 4);
    u32*    Wpc   = (u32*)   alloc((384/2) * 96 * 4);
    u32*    Wpq   = (u32*)   alloc((96/2) * 96 * 4);
    u32*    Wpk   = (u32*)   alloc((96/2) * 96 * 4);
    u32*    Wpv   = (u32*)   alloc((96/2) * 96 * 4);
    u32*    Wpg1  = (u32*)   alloc((96/2) * 288 * 4);
    u32*    Wpg2  = (u32*)   alloc((288/2) * 96 * 4);
    u32*    Wppr  = (u32*)   alloc((96/2) * 192 * 4);
    u32*    WpbQ  = (u32*)   alloc(2 * (192/2) * 192 * 4);
    u32*    WpbK  = (u32*)   alloc(2 * (192/2) * 192 * 4);
    u32*    WpbV  = (u32*)   alloc(2 * (192/2) * 192 * 4);
    u32*    Wpm1  = (u32*)   alloc(2 * (192/2) * 576 * 4);
    u32*    Wpm2  = (u32*)   alloc(2 * (576/2) * 192 * 4);
    half_t* xh    = (half_t*)alloc((size_t)MIN_ * D_ * 2);
    half_t* P     = (half_t*)alloc((size_t)MROWS_ * 384 * 2);     // reused: block q (f32)
    float*  Kin   = (float*) alloc((size_t)MIN_ * D_ * 4);        // reused: block k
    float*  Vin   = (float*) alloc((size_t)MIN_ * D_ * 4);        // reused: block v
    float*  Xo    = (float*) alloc((size_t)MROWS_ * D_ * 4);
    half_t* Xo16  = (half_t*)alloc((size_t)MROWS_ * D_ * 2);
    float*  Qo    = (float*) alloc((size_t)MROWS_ * D_ * 4);
    float*  ncb   = (float*) alloc((size_t)MROWS_ * D_ * 4);
    half_t* Hid16 = (half_t*)alloc((size_t)MROWS_ * BHID_ * 2);
    float*  T1    = (float*) alloc((size_t)MROWS_ * DO_ * 4);
    float*  xb    = (float*) alloc((size_t)MROWS_ * DO_ * 4);
    half_t* xn16  = (half_t*)alloc((size_t)MROWS_ * DO_ * 2);

    // 1) zero the two (mostly-zero) attention maps in d_out
    zero_f<<<4096, 256, 0, stream>>>(out_adown, (size_t)2 * B_ * NOUT_ * NIN_);

    // 2) one-shot weight packing (f32 -> f16-pair dwords)
    repack_conv<<<(384 * 96 + 255) / 256, 256, 0, stream>>>(conv_w, WcT);
    pack_weight<<<(192 * 96 + 255) / 256, 256, 0, stream>>>(WcT, Wpc, 384, 96);
    pack_weight<<<(48 * 96 + 255) / 256, 256, 0, stream>>>(q_w, Wpq, 96, 96);
    pack_weight<<<(48 * 96 + 255) / 256, 256, 0, stream>>>(k_w, Wpk, 96, 96);
    pack_weight<<<(48 * 96 + 255) / 256, 256, 0, stream>>>(v_w, Wpv, 96, 96);
    pack_weight<<<(48 * 288 + 255) / 256, 256, 0, stream>>>(gm1_w, Wpg1, 96, 288);
    pack_weight<<<(144 * 96 + 255) / 256, 256, 0, stream>>>(gm2_w, Wpg2, 288, 96);
    pack_weight<<<(48 * 192 + 255) / 256, 256, 0, stream>>>(proj_w, Wppr, 96, 192);
    for (int i = 0; i < 2; ++i) {
        pack_weight<<<(96 * 192 + 255) / 256, 256, 0, stream>>>(blk_q_w + (size_t)i * 192 * 192, WpbQ + (size_t)i * 96 * 192, 192, 192);
        pack_weight<<<(96 * 192 + 255) / 256, 256, 0, stream>>>(blk_k_w + (size_t)i * 192 * 192, WpbK + (size_t)i * 96 * 192, 192, 192);
        pack_weight<<<(96 * 192 + 255) / 256, 256, 0, stream>>>(blk_v_w + (size_t)i * 192 * 192, WpbV + (size_t)i * 96 * 192, 192, 192);
        pack_weight<<<(96 * 576 + 255) / 256, 256, 0, stream>>>(blk_m1_w + (size_t)i * 192 * 576, Wpm1 + (size_t)i * 96 * 576, 192, 576);
        pack_weight<<<(288 * 192 + 255) / 256, 256, 0, stream>>>(blk_m2_w + (size_t)i * 576 * 192, Wpm2 + (size_t)i * 288 * 192, 576, 192);
    }

    // 3) f16 activations for GEMM A operands
    cvt_half<<<4096, 256, 0, stream>>>(x, xh, (size_t)MIN_ * D_);
    pack_patches<<<(MROWS_ * 384 + 255) / 256, 256, 0, stream>>>(x, P);

    // 4) conv-as-GEMM init projection + LN (keeps f32 Xo + f16 Xo16)
    wmma_gemm<<<dim3(MROWS_ / 128, 2), 256, 0, stream>>>(P, Wpc, conv_b, nullptr, T1, nullptr, MROWS_, 96, 384, 0);
    ln_kernel<96><<<MROWS_ / 8, 256, 0, stream>>>(T1, ln_init_w, ln_init_b, nullptr, Xo, Xo16, MROWS_);

    // 5) K/V projections over all 50176 input tokens
    wmma_gemm<<<dim3(MIN_ / 128, 2), 256, 0, stream>>>(xh, Wpk, k_b, nullptr, Kin, nullptr, MIN_, 96, 96, 0);
    wmma_gemm<<<dim3(MIN_ / 128, 2), 256, 0, stream>>>(xh, Wpv, v_b, nullptr, Vin, nullptr, MIN_, 96, 96, 0);

    // 6) grouping iterations (A_ups only materialized on last iteration)
    for (int it = 0; it < 3; ++it) {
        wmma_gemm<<<dim3(MROWS_ / 128, 2), 256, 0, stream>>>(Xo16, Wpq, q_b, nullptr, Qo, nullptr, MROWS_, 96, 96, 0);
        group_attn<<<MROWS_ / 8, 256, 0, stream>>>(Qo, Kin, Vin, rel_bias, ncb, out_aups, (it == 2) ? 1 : 0);
        ln_kernel<96><<<MROWS_ / 8, 256, 0, stream>>>(ncb, ln_out_w, ln_out_b, Xo, Xo, Xo16, MROWS_);
        wmma_gemm<<<dim3(MROWS_ / 128, HID_ / 48), 256, 0, stream>>>(Xo16, Wpg1, gm1_b, nullptr, nullptr, Hid16, MROWS_, HID_, 96, 1);
        wmma_gemm<<<dim3(MROWS_ / 128, 2), 256, 0, stream>>>(Hid16, Wpg2, gm2_b, nullptr, T1, nullptr, MROWS_, 96, HID_, 0);
        ln_kernel<96><<<MROWS_ / 8, 256, 0, stream>>>(T1, mlp_norm_w, mlp_norm_b, Xo, Xo, Xo16, MROWS_);
    }

    // 7) A_down from final A_ups
    adown_kernel<<<(B_ * NIN_ + 255) / 256, 256, 0, stream>>>(out_aups, out_adown);

    // 8) projection to DO + LN
    wmma_gemm<<<dim3(MROWS_ / 128, DO_ / 48), 256, 0, stream>>>(Xo16, Wppr, proj_b, nullptr, T1, nullptr, MROWS_, DO_, 96, 0);
    ln_kernel<192><<<MROWS_ / 8, 256, 0, stream>>>(T1, norm_w, norm_b, nullptr, xb, nullptr, MROWS_);

    // 9) SeNaTra blocks (reuse big grouping buffers for q/k/v f32)
    float* qb = (float*)P; float* kb = Kin; float* vb = Vin;
    for (int i = 0; i < 2; ++i) {
        size_t bo = (size_t)i * DO_, m1bo = (size_t)i * BHID_;
        const u32* wq = WpbQ + (size_t)i * 96 * 192;
        const u32* wk = WpbK + (size_t)i * 96 * 192;
        const u32* wv = WpbV + (size_t)i * 96 * 192;
        const u32* wm1 = Wpm1 + (size_t)i * 96 * 576;
        const u32* wm2 = Wpm2 + (size_t)i * 288 * 192;
        ln_kernel<192><<<MROWS_ / 8, 256, 0, stream>>>(xb, blk_ln1_w + bo, blk_ln1_b + bo, nullptr, nullptr, xn16, MROWS_);
        wmma_gemm<<<dim3(MROWS_ / 128, 4), 256, 0, stream>>>(xn16, wq, blk_q_b + bo, nullptr, qb, nullptr, MROWS_, DO_, DO_, 0);
        wmma_gemm<<<dim3(MROWS_ / 128, 4), 256, 0, stream>>>(xn16, wk, blk_k_b + bo, nullptr, kb, nullptr, MROWS_, DO_, DO_, 0);
        wmma_gemm<<<dim3(MROWS_ / 128, 4), 256, 0, stream>>>(xn16, wv, blk_v_b + bo, nullptr, vb, nullptr, MROWS_, DO_, DO_, 0);
        win_attn<<<MROWS_ / 8, 256, 0, stream>>>(qb, kb, vb, xb, xb);
        ln_kernel<192><<<MROWS_ / 8, 256, 0, stream>>>(xb, blk_ln2_w + bo, blk_ln2_b + bo, nullptr, nullptr, xn16, MROWS_);
        wmma_gemm<<<dim3(MROWS_ / 128, BHID_ / 48), 256, 0, stream>>>(xn16, wm1, blk_m1_b + m1bo, nullptr, nullptr, Hid16, MROWS_, BHID_, DO_, 1);
        float* dst = (i == 1) ? out_xb : xb;
        wmma_gemm<<<dim3(MROWS_ / 128, 4), 256, 0, stream>>>(Hid16, wm2, blk_m2_b + bo, xb, dst, nullptr, MROWS_, DO_, BHID_, 0);
    }
    (void)in_sizes; (void)n_in; (void)out_size; (void)ws_size;
}
